// CoordinationMemory_71494025609991
// MI455X (gfx1250) — matrix operand order repaired and checked by reading, not compiled, not run on hardware
//
#include <hip/hip_runtime.h>
#include <stdint.h>

typedef __attribute__((ext_vector_type(16))) __bf16 v16bf;
typedef __attribute__((ext_vector_type(8)))  float  v8f;
typedef __attribute__((ext_vector_type(4)))  float  f4;   // native vec4 (works with nontemporal builtins)

__device__ __forceinline__ unsigned short f32_to_bf16(float f) {
  union { float f; unsigned int u; } c; c.f = f;
  unsigned int u = c.u;
  u += 0x7FFFu + ((u >> 16) & 1u);          // round-to-nearest-even
  return (unsigned short)(u >> 16);
}

// ---------------------------------------------------------------------------
// Kernel 1: streaming copy memory -> out (512 MB each way, the real cost)
// ---------------------------------------------------------------------------
__global__ void copy_mem_kernel(const f4* __restrict__ src,
                                f4* __restrict__ dst, int n4) {
  int i = blockIdx.x * blockDim.x + threadIdx.x;
  int stride = gridDim.x * blockDim.x;
  for (; i < n4; i += stride) {
    f4 v = __builtin_nontemporal_load(&src[i]);
    __builtin_nontemporal_store(v, &dst[i]);
  }
}

// ---------------------------------------------------------------------------
// Kernel 2: fused GEMM (M=4096,K=1024,N=256 via bf16 WMMA) + bias + LayerNorm
//           + tanh + scatter into out[n, idx[n], :]
// Block: 512 threads = 16 waves. Each block: 16 rows x 256 cols.
// Each wave: one 16x16 output tile, K-loop of 32 wmma_f32_16x16x32_bf16.
// ---------------------------------------------------------------------------
__global__ __launch_bounds__(512)
void update_kernel(const float* __restrict__ memory,
                   const long long* __restrict__ veh_idx,
                   const float* __restrict__ veh_repr,
                   const float* __restrict__ cust_repr,
                   const float* __restrict__ edge_emb,
                   const float* __restrict__ W_in,
                   const float* __restrict__ b_in,
                   const float* __restrict__ W_h,
                   const float* __restrict__ b_h,
                   const float* __restrict__ gamma,
                   const float* __restrict__ beta,
                   float* __restrict__ out) {
  __shared__ unsigned short sA[16][1032];   // bf16 bits, A rows, padded (33 KB)
  __shared__ unsigned short sBt[256][40];   // bf16 bits, Bt[col][k-in-step] (20 KB)
  __shared__ float sSum[16];
  __shared__ float sSumSq[16];
  __shared__ int   sIdx[16];

  const int tid    = threadIdx.x;
  const int wave   = tid >> 5;
  const int lane   = tid & 31;
  const int row0   = blockIdx.x * 16;

  if (tid < 16) {
    sSum[tid]   = 0.0f;
    sSumSq[tid] = 0.0f;
    sIdx[tid]   = (int)veh_idx[row0 + tid];
  }
  __syncthreads();

  // ---- Stage A (16 rows x 1024 K) as bf16 into LDS: thread = (row=wave, 32 k's)
  {
    const int m  = wave;            // 0..15
    const int k0 = lane * 32;       // 0..992, always within one source region
    const int n  = row0 + m;
    const float* src;
    if (k0 < 256)       src = veh_repr  + n * 256 + k0;
    else if (k0 < 512)  src = cust_repr + n * 256 + (k0 - 256);
    else if (k0 < 768)  src = edge_emb  + n * 256 + (k0 - 512);
    else                src = memory + (size_t)(n * 128 + sIdx[m]) * 256 + (k0 - 768);
#pragma unroll
    for (int j = 0; j < 32; j += 4) {
      f4 v = *(const f4*)(src + j);
      sA[m][k0 + j + 0] = f32_to_bf16(v.x);
      sA[m][k0 + j + 1] = f32_to_bf16(v.y);
      sA[m][k0 + j + 2] = f32_to_bf16(v.z);
      sA[m][k0 + j + 3] = f32_to_bf16(v.w);
    }
  }

  const int colw = lane & 15;              // col within wave's 16-wide tile
  const int g    = lane >> 4;              // half-wave select
  const int colg = wave * 16 + colw;       // global output column 0..255

  v8f acc = {};

  for (int s = 0; s < 32; ++s) {
    __syncthreads();   // covers A-staging (s==0) and sBt reuse (s>0)
    // ---- Stage Bt for K-step s: W rows [s*32, s*32+32), transposed into LDS
    {
      const int kk = tid >> 4;             // 0..31
      const int c0 = (tid & 15) * 16;      // 0..240
      const int r  = s * 32 + kk;
      const float* wsrc = (r < 768) ? (W_in + r * 256 + c0)
                                    : (W_h + (r - 768) * 256 + c0);
#pragma unroll
      for (int j = 0; j < 16; j += 4) {
        f4 v = *(const f4*)(wsrc + j);
        sBt[c0 + j + 0][kk] = f32_to_bf16(v.x);
        sBt[c0 + j + 1][kk] = f32_to_bf16(v.y);
        sBt[c0 + j + 2][kk] = f32_to_bf16(v.z);
        sBt[c0 + j + 3][kk] = f32_to_bf16(v.w);
      }
    }
    __syncthreads();

    // ---- Load fragments, contiguous k order (k = 16*g + e) for both A and B
    union { v16bf v; unsigned short u[16]; } a, b;
    const int m = lane & 15;
    // A: lane half g holds K = s*32 + g*16 + {0..15}
    const unsigned short* pa = &sA[m][s * 32 + g * 16];
#pragma unroll
    for (int e = 0; e < 16; ++e) { a.u[e] = pa[e]; }
    // B: lane half g holds K = g*16 + {0..15} for column colg
    const unsigned short* pb = &sBt[colg][g * 16];
#pragma unroll
    for (int e = 0; e < 16; ++e) { b.u[e] = pb[e]; }

    acc = __builtin_amdgcn_wmma_f32_16x16x32_bf16(false, a.v, false, b.v,
                                                  (short)0, acc, false, false);
  }

  // ---- bias, then per-row mean / meansq reduction (row = r + 8*g)
  const float bias = b_in[colg] + b_h[colg];
  float hv[8];
#pragma unroll
  for (int r = 0; r < 8; ++r) hv[r] = acc[r] + bias;

#pragma unroll
  for (int r = 0; r < 8; ++r) {
    float sv = hv[r];
    float sq = hv[r] * hv[r];
#pragma unroll
    for (int mask = 1; mask < 16; mask <<= 1) {   // reduce within 16-lane half
      sv += __shfl_xor(sv, mask, 32);
      sq += __shfl_xor(sq, mask, 32);
    }
    if (colw == 0) {
      atomicAdd(&sSum[r + 8 * g], sv);
      atomicAdd(&sSumSq[r + 8 * g], sq);
    }
  }
  __syncthreads();

  // ---- LayerNorm + tanh + scatter to out[n, idx[n], colg]
  const float gam = gamma[colg];
  const float bet = beta[colg];
#pragma unroll
  for (int r = 0; r < 8; ++r) {
    const int   rl  = r + 8 * g;
    const float mu  = sSum[rl]   * (1.0f / 256.0f);
    const float var = sSumSq[rl] * (1.0f / 256.0f) - mu * mu;
    const float rs  = rsqrtf(var + 1e-5f);
    const float ln  = (hv[r] - mu) * rs * gam + bet;
    const float res = tanhf(ln);
    const int   n   = row0 + rl;
    out[((size_t)n * 128 + sIdx[rl]) * 256 + colg] = res;
  }
}

// ---------------------------------------------------------------------------
extern "C" void kernel_launch(void* const* d_in, const int* in_sizes, int n_in,
                              void* d_out, int out_size, void* d_ws, size_t ws_size,
                              hipStream_t stream) {
  const float*     memory_p  = (const float*)d_in[0];
  const long long* veh_idx   = (const long long*)d_in[1];   // int64 in reference
  const float*     veh_repr  = (const float*)d_in[2];
  const float*     cust_repr = (const float*)d_in[3];
  const float*     edge_emb  = (const float*)d_in[4];
  const float*     W_in      = (const float*)d_in[5];
  const float*     b_in      = (const float*)d_in[6];
  const float*     W_h       = (const float*)d_in[7];
  const float*     b_h       = (const float*)d_in[8];
  const float*     gamma_p   = (const float*)d_in[9];
  const float*     beta_p    = (const float*)d_in[10];
  float*           out       = (float*)d_out;

  // 1) bulk copy memory -> out (134,217,728 floats = 33,554,432 float4)
  const int n4 = (4096 * 128 * 256) / 4;
  copy_mem_kernel<<<4096, 256, 0, stream>>>((const f4*)memory_p,
                                            (f4*)out, n4);

  // 2) fused GEMM (WMMA bf16) + LN + tanh + scatter (256 row-blocks of 16)
  update_kernel<<<256, 512, 0, stream>>>(memory_p, veh_idx, veh_repr, cust_repr,
                                         edge_emb, W_in, b_in, W_h, b_h,
                                         gamma_p, beta_p, out);
}
